// GNN_57080115364690
// MI455X (gfx1250) — matrix operand order repaired and checked by reading, not compiled, not run on hardware
//
#include <hip/hip_runtime.h>
#include <hip/hip_bf16.h>

// ---------------------------------------------------------------------------
// GIN + BiLSTM graph embedding, CDNA5 (gfx1250) WMMA implementation.
// B=128 graphs, N=256 nodes, D_IN=512, H=1024, L=5 layers.
// All GEMMs run on v_wmma_f32_16x16x32_bf16 (bf16 in, f32 accumulate).
// A-tiles staged with GLOBAL_LOAD_ASYNC_TO_LDS_B128 when the builtin exists.
// ---------------------------------------------------------------------------

#define GB   128      // batch (graphs)
#define GN   256      // nodes per graph
#define GDIN 512
#define GH   1024
#define GL   5
#define BNROWS (GB * GN)   // 32768

typedef __bf16 bf16_t;
typedef __attribute__((ext_vector_type(16))) __bf16 v16bf;
typedef __attribute__((ext_vector_type(8)))  __bf16 v8bf;
typedef __attribute__((ext_vector_type(8)))  float  v8f;
typedef __attribute__((ext_vector_type(4)))  unsigned int u32x4;
typedef __attribute__((ext_vector_type(4)))  int       v4i;

#define LDSS 48   // LDS row stride in bf16 elems (32 data + 16 pad) -> 96 B, 16B aligned

// Async memory->LDS DMA path (gfx1250 GLOBAL_LOAD_ASYNC_TO_LDS_B128, ASYNCcnt).
#if defined(__gfx1250__) && \
    __has_builtin(__builtin_amdgcn_global_load_async_to_lds_b128) && \
    __has_builtin(__builtin_amdgcn_s_wait_asynccnt)
#define USE_ASYNC_LDS 1
#else
#define USE_ASYNC_LDS 0
#endif

// ---------------------------------------------------------------------------
// Generic batched GEMM: C[b] = act( A[b] (MxK) * B[b] (KxN) + bias + scale*resid )
// A,B,resid bf16 row-major; C written as bf16 (OUTB) and/or f32 (OUTF).
// Workgroup: 256 threads (8 waves), tile 128(M) x 64(N), K stepped by 32.
// Each wave computes a 32x32 tile = 2x2 v_wmma_f32_16x16x32_bf16.
// Compile-time flags keep the epilogue branch-free.
// ---------------------------------------------------------------------------
template <bool BIAS, bool RES, bool RELU, bool OUTB, bool OUTF>
__global__ __launch_bounds__(256)
void gemm_bf16_wmma(const bf16_t* __restrict__ A, const bf16_t* __restrict__ B,
                    const float* __restrict__ bias, const bf16_t* __restrict__ resid,
                    const float* __restrict__ epsP,
                    bf16_t* __restrict__ Cb, float* __restrict__ Cf,
                    int M, int N, int K,
                    long long sA, long long sB, long long sC)
{
    __shared__ __attribute__((aligned(16))) bf16_t ldsA[128 * LDSS]; // [M][K] row-major
    __shared__ __attribute__((aligned(16))) bf16_t ldsB[ 64 * LDSS]; // [N][K] (transposed)

    const int tid  = threadIdx.x;
    const int lane = tid & 31;
    const int wave = tid >> 5;
    const int wr   = wave & 3;        // wave's 32-row slot in the 128-row tile
    const int wc   = wave >> 2;       // wave's 32-col slot in the 64-col tile
    const int hi   = lane >> 4;       // lane half (selects K sub-ranges per ISA layout)
    const int mr   = lane & 15;       // M (or N) index within fragment

    const long long batch = blockIdx.z;
    const bf16_t* Ab = A + batch * sA;
    const bf16_t* Bb = B + batch * sB;
    const int baseM = blockIdx.y * 128;
    const int baseN = blockIdx.x * 64;

    v8f acc[2][2] = {};

    for (int k0 = 0; k0 < K; k0 += 32) {
        // ---- stage A tile 128x32 into LDS (512 16-byte chunks / 256 threads) ----
#pragma unroll
        for (int c = 0; c < 2; ++c) {
            int chunk = tid + c * 256;
            int row = chunk >> 2, part = chunk & 3;
            const bf16_t* gsrc = Ab + (size_t)(baseM + row) * K + k0 + part * 8;
            bf16_t*       ldst = ldsA + row * LDSS + part * 8;
#if USE_ASYNC_LDS
            __builtin_amdgcn_global_load_async_to_lds_b128(
                (__attribute__((address_space(1))) v4i*)gsrc,
                (__attribute__((address_space(3))) v4i*)ldst, 0, 0);
#else
            *(u32x4*)ldst = *(const u32x4*)gsrc;
#endif
        }
        // ---- stage B tile 32x64, transposed into LDS as [N][K] ----
        {
            int kr = tid >> 3;                  // 0..31
            int nl = (tid & 7) * 8;             // 0..56
            u32x4 raw = *(const u32x4*)(Bb + (size_t)(k0 + kr) * N + baseN + nl);
            const bf16_t* tv = (const bf16_t*)&raw;
#pragma unroll
            for (int j = 0; j < 8; ++j)
                ldsB[(nl + j) * LDSS + kr] = tv[j];
        }
#if USE_ASYNC_LDS
        __builtin_amdgcn_s_wait_asynccnt(0);
#endif
        __syncthreads();

        // ---- fragments (ISA 7.12.2 16-bit A 16x32 layout; B mirrored on N) ----
        v16bf af[2], bfr[2];
#pragma unroll
        for (int i = 0; i < 2; ++i) {
            const bf16_t* ap = ldsA + (wr * 32 + i * 16 + mr) * LDSS;
            ((v8bf*)&af[i])[0] = *(const v8bf*)(ap + hi * 8);        // K = 0..7  (+8 hi)
            ((v8bf*)&af[i])[1] = *(const v8bf*)(ap + 16 + hi * 8);   // K = 16..23(+8 hi)
        }
#pragma unroll
        for (int j = 0; j < 2; ++j) {
            const bf16_t* bp = ldsB + (wc * 32 + j * 16 + mr) * LDSS + hi * 8;
            ((v8bf*)&bfr[j])[0] = *(const v8bf*)(bp);
            ((v8bf*)&bfr[j])[1] = *(const v8bf*)(bp + 16);
        }
#pragma unroll
        for (int i = 0; i < 2; ++i)
#pragma unroll
            for (int j = 0; j < 2; ++j)
                acc[i][j] = __builtin_amdgcn_wmma_f32_16x16x32_bf16(
                    false, af[i], false, bfr[j], (short)0, acc[i][j], false, false);
        __syncthreads();
    }

    // ---- epilogue: bias + residual + relu, write bf16/f32 (branch-free) ----
    float scale = 0.0f;
    if constexpr (RES) scale = 1.0f + *epsP;
#pragma unroll
    for (int i = 0; i < 2; ++i)
#pragma unroll
        for (int j = 0; j < 2; ++j)
#pragma unroll
            for (int r = 0; r < 8; ++r) {
                int m = baseM + wr * 32 + i * 16 + hi * 8 + r;
                int n = baseN + wc * 32 + j * 16 + mr;
                size_t idx = (size_t)(batch * sC) + (size_t)m * N + n;
                float v = acc[i][j][r];
                if constexpr (BIAS) v += bias[n];
                if constexpr (RES)  v += scale * (float)resid[idx];
                if constexpr (RELU) v = fmaxf(v, 0.0f);
                if constexpr (OUTB) Cb[idx] = (bf16_t)v;
                if constexpr (OUTF) Cf[idx] = v;
            }
}

// ---------------------------------------------------------------------------
// Helpers
// ---------------------------------------------------------------------------
__global__ void f32_to_bf16(const float* __restrict__ in, bf16_t* __restrict__ out,
                            long long n)
{
    long long i = (long long)blockIdx.x * blockDim.x + threadIdx.x;
    long long stride = (long long)gridDim.x * blockDim.x;
    for (; i < n; i += stride) out[i] = (bf16_t)in[i];
}

// in [R,C] row-major -> out [C,R] row-major (bf16)
__global__ void f32_to_bf16_T(const float* __restrict__ in, bf16_t* __restrict__ out,
                              int R, int C)
{
    long long n = (long long)R * C;
    long long i = (long long)blockIdx.x * blockDim.x + threadIdx.x;
    long long stride = (long long)gridDim.x * blockDim.x;
    for (; i < n; i += stride) {
        int r = (int)(i / C), c = (int)(i % C);
        out[(size_t)c * R + r] = (bf16_t)in[i];
    }
}

// max over nodes: h [GB*GN, GH] bf16 -> seq slot [GB, GL, GH] (bf16 + f32)
__global__ void rowmax_kernel(const bf16_t* __restrict__ h, bf16_t* __restrict__ seqB,
                              float* __restrict__ seqF, int slot)
{
    int b = blockIdx.y;
    int k = blockIdx.x * blockDim.x + threadIdx.x;   // feature, < GH
    const bf16_t* hp = h + (size_t)b * GN * GH + k;
    float m = -1e30f;
    for (int n = 0; n < GN; ++n) m = fmaxf(m, (float)hp[(size_t)n * GH]);
    size_t o = ((size_t)b * GL + slot) * GH + k;
    seqB[o] = (bf16_t)m;
    seqF[o] = m;
}

__global__ void lstm_init(bf16_t* __restrict__ h, float* __restrict__ c, int n)
{
    int i = blockIdx.x * blockDim.x + threadIdx.x;
    if (i < n) { h[i] = (bf16_t)0.0f; c[i] = 0.0f; }
}

// one LSTM step (torch gate order i,f,g,o). gx already offset to timestep.
__global__ void lstm_cell(const float* __restrict__ gx, int gxStride,
                          const float* __restrict__ gh,
                          float* __restrict__ c, bf16_t* __restrict__ hbf,
                          float* __restrict__ hout, int houtStride)
{
    int idx = blockIdx.x * blockDim.x + threadIdx.x;
    if (idx >= GB * GH) return;
    int b = idx / GH, k = idx % GH;
    const float* gxr = gx + (size_t)b * gxStride;
    const float* ghr = gh + (size_t)b * 4 * GH;
    float gi = gxr[k]          + ghr[k];
    float gf = gxr[k +     GH] + ghr[k +     GH];
    float gg = gxr[k + 2 * GH] + ghr[k + 2 * GH];
    float go = gxr[k + 3 * GH] + ghr[k + 3 * GH];
    float si = 1.0f / (1.0f + __expf(-gi));
    float sf = 1.0f / (1.0f + __expf(-gf));
    float so = 1.0f / (1.0f + __expf(-go));
    float cn = sf * c[idx] + si * tanhf(gg);
    c[idx] = cn;
    float hv = so * tanhf(cn);
    hbf[idx] = (bf16_t)hv;
    hout[(size_t)b * houtStride + k] = hv;
}

// softmax over the L layer axis, then weighted sum with seq -> [GB, GH] f32
__global__ void softmax_sum(const float* __restrict__ logits,
                            const float* __restrict__ seqF,
                            float* __restrict__ out)
{
    int idx = blockIdx.x * blockDim.x + threadIdx.x;
    if (idx >= GB * GH) return;
    int b = idx / GH, k = idx % GH;
    float v[GL], mx = -1e30f;
#pragma unroll
    for (int t = 0; t < GL; ++t) {
        v[t] = logits[((size_t)b * GL + t) * GH + k];
        mx = fmaxf(mx, v[t]);
    }
    float s = 0.0f;
#pragma unroll
    for (int t = 0; t < GL; ++t) { v[t] = __expf(v[t] - mx); s += v[t]; }
    float r = 0.0f;
#pragma unroll
    for (int t = 0; t < GL; ++t)
        r += (v[t] / s) * seqF[((size_t)b * GL + t) * GH + k];
    out[idx] = r;
}

// ---------------------------------------------------------------------------
// Host orchestration
// ---------------------------------------------------------------------------
extern "C" void kernel_launch(void* const* d_in, const int* in_sizes, int n_in,
                              void* d_out, int out_size, void* d_ws, size_t ws_size,
                              hipStream_t stream)
{
    (void)in_sizes; (void)n_in; (void)out_size; (void)ws_size;

    const float* features = (const float*)d_in[0];   // [128,256,512]
    const float* adj      = (const float*)d_in[1];   // [128,256,256]
    const float* eps      = (const float*)d_in[2];   // [4]
    const float* m0w1     = (const float*)d_in[3];   // [512,1024]
    const float* m0b1     = (const float*)d_in[4];
    const float* m0w2     = (const float*)d_in[5];   // [1024,1024]
    const float* m0b2     = (const float*)d_in[6];
    const float* mlw1     = (const float*)d_in[7];   // [4,1024,1024]
    const float* mlb1     = (const float*)d_in[8];   // [4,1024]
    const float* mlw2     = (const float*)d_in[9];
    const float* mlb2     = (const float*)d_in[10];
    const float* wih_f    = (const float*)d_in[11];  // [4096,1024]
    const float* whh_f    = (const float*)d_in[12];
    const float* b_f      = (const float*)d_in[13];  // [4096]
    const float* wih_b    = (const float*)d_in[14];
    const float* whh_b    = (const float*)d_in[15];
    const float* b_b      = (const float*)d_in[16];
    const float* final_w  = (const float*)d_in[17];  // [2048,1024]
    const float* final_b  = (const float*)d_in[18];  // [1024]
    float* out = (float*)d_out;                      // [128,1024]

    // ---- bump allocator over d_ws (~315 MB total) ----
    char* base = (char*)d_ws;
    size_t off = 0;
    auto alloc = [&](size_t bytes) -> void* {
        off = (off + 255) & ~(size_t)255;
        void* p = base + off;
        off += bytes;
        return p;
    };
    bf16_t* WA   = (bf16_t*)alloc((size_t)BNROWS * GH * 2);      // h (bf16)
    bf16_t* WB   = (bf16_t*)alloc((size_t)BNROWS * GH * 2);      // mlp hidden
    bf16_t* WC   = (bf16_t*)alloc((size_t)BNROWS * GH * 2);      // features / agg
    bf16_t* adjB = (bf16_t*)alloc((size_t)GB * GN * GN * 2);
    bf16_t* w01  = (bf16_t*)alloc((size_t)GDIN * GH * 2);
    bf16_t* w02  = (bf16_t*)alloc((size_t)GH * GH * 2);
    bf16_t* wl1  = (bf16_t*)alloc((size_t)4 * GH * GH * 2);
    bf16_t* wl2  = (bf16_t*)alloc((size_t)4 * GH * GH * 2);
    bf16_t* wihTf= (bf16_t*)alloc((size_t)GH * 4 * GH * 2);      // [1024,4096]
    bf16_t* whhTf= (bf16_t*)alloc((size_t)GH * 4 * GH * 2);
    bf16_t* wihTb= (bf16_t*)alloc((size_t)GH * 4 * GH * 2);
    bf16_t* whhTb= (bf16_t*)alloc((size_t)GH * 4 * GH * 2);
    bf16_t* finW = (bf16_t*)alloc((size_t)2 * GH * GH * 2);      // [2048,1024]
    bf16_t* seqB = (bf16_t*)alloc((size_t)GB * GL * GH * 2);
    float*  seqF = (float*) alloc((size_t)GB * GL * GH * 4);
    float*  gxF  = (float*) alloc((size_t)GB * GL * 4 * GH * 4); // [640,4096]
    float*  gxB  = (float*) alloc((size_t)GB * GL * 4 * GH * 4);
    float*  ghBuf= (float*) alloc((size_t)GB * 4 * GH * 4);      // [128,4096]
    bf16_t* hF   = (bf16_t*)alloc((size_t)GB * GH * 2);
    bf16_t* hB   = (bf16_t*)alloc((size_t)GB * GH * 2);
    float*  cF   = (float*) alloc((size_t)GB * GH * 4);
    float*  cB   = (float*) alloc((size_t)GB * GH * 4);
    float*  outF = (float*) alloc((size_t)GB * GL * 2 * GH * 4); // [640,2048]
    bf16_t* outFb= (bf16_t*)alloc((size_t)GB * GL * 2 * GH * 2);
    float*  logit= (float*) alloc((size_t)GB * GL * GH * 4);     // [640,1024]

    auto CVT = [&](const float* s, bf16_t* d, long long n) {
        hipLaunchKernelGGL(f32_to_bf16, dim3(2048), dim3(256), 0, stream, s, d, n);
    };

    // ---- 1. precision conversion of inputs/weights ----
    CVT(features, WC, (long long)BNROWS * GDIN);
    CVT(adj, adjB, (long long)GB * GN * GN);
    CVT(m0w1, w01, (long long)GDIN * GH);
    CVT(m0w2, w02, (long long)GH * GH);
    CVT(mlw1, wl1, (long long)4 * GH * GH);
    CVT(mlw2, wl2, (long long)4 * GH * GH);
    CVT(final_w, finW, (long long)2 * GH * GH);
    hipLaunchKernelGGL(f32_to_bf16_T, dim3(4096), dim3(256), 0, stream, wih_f, wihTf, 4 * GH, GH);
    hipLaunchKernelGGL(f32_to_bf16_T, dim3(4096), dim3(256), 0, stream, whh_f, whhTf, 4 * GH, GH);
    hipLaunchKernelGGL(f32_to_bf16_T, dim3(4096), dim3(256), 0, stream, wih_b, wihTb, 4 * GH, GH);
    hipLaunchKernelGGL(f32_to_bf16_T, dim3(4096), dim3(256), 0, stream, whh_b, whhTb, 4 * GH, GH);

    // ---- 2. MLP0: h = relu(x@w1+b1)@w2+b2 ----
    gemm_bf16_wmma<true, false, true, true, false>
        <<<dim3(GH / 64, BNROWS / 128, 1), 256, 0, stream>>>(
            WC, w01, m0b1, nullptr, nullptr, WB, nullptr, BNROWS, GH, GDIN, 0, 0, 0);
    gemm_bf16_wmma<true, false, false, true, false>
        <<<dim3(GH / 64, BNROWS / 128, 1), 256, 0, stream>>>(
            WB, w02, m0b2, nullptr, nullptr, WA, nullptr, BNROWS, GH, GH, 0, 0, 0);
    hipLaunchKernelGGL(rowmax_kernel, dim3(GH / 256, GB), dim3(256), 0, stream, WA, seqB, seqF, 0);

    // ---- 3. GIN layers: h = MLP(adj@h + (1+eps)h) ----
    for (int i = 0; i < GL - 1; ++i) {
        gemm_bf16_wmma<false, true, false, true, false>
            <<<dim3(GH / 64, GN / 128, GB), 256, 0, stream>>>(
                adjB, WA, nullptr, WA, eps + i, WC, nullptr, GN, GH, GN,
                (long long)GN * GN, (long long)GN * GH, (long long)GN * GH);
        gemm_bf16_wmma<true, false, true, true, false>
            <<<dim3(GH / 64, BNROWS / 128, 1), 256, 0, stream>>>(
                WC, wl1 + (size_t)i * GH * GH, mlb1 + i * GH, nullptr, nullptr,
                WB, nullptr, BNROWS, GH, GH, 0, 0, 0);
        gemm_bf16_wmma<true, false, false, true, false>
            <<<dim3(GH / 64, BNROWS / 128, 1), 256, 0, stream>>>(
                WB, wl2 + (size_t)i * GH * GH, mlb2 + i * GH, nullptr, nullptr,
                WA, nullptr, BNROWS, GH, GH, 0, 0, 0);
        hipLaunchKernelGGL(rowmax_kernel, dim3(GH / 256, GB), dim3(256), 0, stream,
                           WA, seqB, seqF, i + 1);
    }

    // ---- 4. BiLSTM over the L=5 graph features ----
    // precompute x@W_ih^T + b for all timesteps of both directions
    gemm_bf16_wmma<true, false, false, false, true>
        <<<dim3(4 * GH / 64, GB * GL / 128, 1), 256, 0, stream>>>(
            seqB, wihTf, b_f, nullptr, nullptr, nullptr, gxF, GB * GL, 4 * GH, GH, 0, 0, 0);
    gemm_bf16_wmma<true, false, false, false, true>
        <<<dim3(4 * GH / 64, GB * GL / 128, 1), 256, 0, stream>>>(
            seqB, wihTb, b_b, nullptr, nullptr, nullptr, gxB, GB * GL, 4 * GH, GH, 0, 0, 0);

    hipLaunchKernelGGL(lstm_init, dim3((GB * GH + 255) / 256), dim3(256), 0, stream, hF, cF, GB * GH);
    for (int t = 0; t < GL; ++t) {
        gemm_bf16_wmma<false, false, false, false, true>
            <<<dim3(4 * GH / 64, GB / 128, 1), 256, 0, stream>>>(
                hF, whhTf, nullptr, nullptr, nullptr, nullptr, ghBuf, GB, 4 * GH, GH, 0, 0, 0);
        hipLaunchKernelGGL(lstm_cell, dim3((GB * GH + 255) / 256), dim3(256), 0, stream,
                           gxF + (size_t)t * 4 * GH, GL * 4 * GH, ghBuf, cF, hF,
                           outF + (size_t)t * 2 * GH, GL * 2 * GH);
    }
    hipLaunchKernelGGL(lstm_init, dim3((GB * GH + 255) / 256), dim3(256), 0, stream, hB, cB, GB * GH);
    for (int t = 0; t < GL; ++t) {
        int tt = GL - 1 - t;   // backward direction walks the sequence reversed
        gemm_bf16_wmma<false, false, false, false, true>
            <<<dim3(4 * GH / 64, GB / 128, 1), 256, 0, stream>>>(
                hB, whhTb, nullptr, nullptr, nullptr, nullptr, ghBuf, GB, 4 * GH, GH, 0, 0, 0);
        hipLaunchKernelGGL(lstm_cell, dim3((GB * GH + 255) / 256), dim3(256), 0, stream,
                           gxB + (size_t)tt * 4 * GH, GL * 4 * GH, ghBuf, cB, hB,
                           outF + (size_t)tt * 2 * GH + GH, GL * 2 * GH);
    }

    // ---- 5. attention over layers + weighted sum ----
    CVT(outF, outFb, (long long)GB * GL * 2 * GH);
    gemm_bf16_wmma<true, false, false, false, true>
        <<<dim3(GH / 64, GB * GL / 128, 1), 256, 0, stream>>>(
            outFb, finW, final_b, nullptr, nullptr, nullptr, logit,
            GB * GL, GH, 2 * GH, 0, 0, 0);
    hipLaunchKernelGGL(softmax_sum, dim3((GB * GH + 255) / 256), dim3(256), 0, stream,
                       logit, seqF, out);
}
